// Attention_68169720922266
// MI455X (gfx1250) — matrix operand order repaired and checked by reading, not compiled, not run on hardware
//
#include <hip/hip_runtime.h>
#include <hip/hip_bf16.h>
#include <math.h>

// ---------------------------------------------------------------------------
// Types matching gfx1250 WMMA builtins
// ---------------------------------------------------------------------------
typedef __bf16 bf16_t;
typedef bf16_t v16bf __attribute__((ext_vector_type(16)));
typedef bf16_t v8bf  __attribute__((ext_vector_type(8)));
typedef float  v8f   __attribute__((ext_vector_type(8)));

static __device__ __forceinline__ bf16_t f2bf(float f) {
    unsigned u = __float_as_uint(f);
    unsigned r = (u + 0x7FFFu + ((u >> 16) & 1u)) >> 16;   // round-to-nearest-even
    unsigned short s = (unsigned short)r;
    bf16_t b;
    __builtin_memcpy(&b, &s, 2);
    return b;
}
static __device__ __forceinline__ float bf2f(bf16_t b) {
    unsigned short s;
    __builtin_memcpy(&s, &b, 2);
    return __uint_as_float(((unsigned)s) << 16);
}

// A-fragment (16x32 bf16, M x K): lane<16 holds row=lane, K {0..15};
// lane>=16 holds row=lane-16, K {16..31}. Caller passes base pointer
// already offset by row*stride + koff; chunks are [0..7] and [8..15].
static __device__ __forceinline__ v16bf load_afrag(const bf16_t* p) {
    v8bf lo = *(const v8bf*)p;
    v8bf hi = *(const v8bf*)(p + 8);
    return __builtin_shufflevector(lo, hi, 0,1,2,3,4,5,6,7,8,9,10,11,12,13,14,15);
}

static __device__ __forceinline__ v8f wmma_bf16(v16bf a, v16bf b, v8f c) {
    return __builtin_amdgcn_wmma_f32_16x16x32_bf16(false, a, false, b, (short)0, c,
                                                   false, false);
}

// tanh(x) = 1 - 2/(exp(2x)+1)  (cheap, uses fast exp; avoids libm expansion)
static __device__ __forceinline__ float fast_tanh(float x) {
    float t = __expf(2.0f * x);
    return 1.0f - 2.0f / (t + 1.0f);
}

// ---------------------------------------------------------------------------
// Problem constants
// ---------------------------------------------------------------------------
#define HEADS   8
#define DHH     64      // per-head half dim
#define DV      128     // per-head value dim (2*DHH)
#define DIMC    1024
#define BATCH   2
#define SEQ     2048
#define MTOT    (BATCH * SEQ)   // 4096
#define LAMBDA_INIT 0.5f
// |logits| bound: cos_scale <= 15*1.5 = 22.5, |cos| <= 1, |gate| <= 1  ->  |S| < 24
#define LOGIT_CAP 24.0f

// ---------------------------------------------------------------------------
// Elementwise f32 -> bf16 conversion
// ---------------------------------------------------------------------------
__global__ void f32_to_bf16_kernel(const float* __restrict__ src,
                                   bf16_t* __restrict__ dst, int n) {
    int i = blockIdx.x * blockDim.x + threadIdx.x;
    if (i < n) dst[i] = f2bf(src[i]);
}

// ---------------------------------------------------------------------------
// bf16 GEMM: C[M][N] = A[M][K] * B[N][K]^T  (B row-major over [N][K])
// Block: 256 thr (8 waves), tile 128x128, K-step 32.
// Wave (wm 0..3, wn 0..1) owns a 32x64 region = 2x4 WMMA tiles.
// ---------------------------------------------------------------------------
template <bool F32OUT>
__global__ __launch_bounds__(256)
void gemm_kernel(const bf16_t* __restrict__ A, const bf16_t* __restrict__ Bm,
                 void* __restrict__ C, const float* __restrict__ bias,
                 int M, int Nn, int K) {
    __shared__ alignas(64) bf16_t As[128 * 32];
    __shared__ alignas(64) bf16_t Bs[128 * 32];

    const int tid  = threadIdx.x;
    const int wave = tid >> 5;
    const int lane = tid & 31;
    const int l15  = lane & 15;
    const int hi16 = lane >> 4;
    const int wm   = wave >> 1;
    const int wn   = wave & 1;
    const int blockN = blockIdx.x * 128;
    const int blockM = blockIdx.y * 128;

    v8f acc[2][4];
#pragma unroll
    for (int mt = 0; mt < 2; ++mt)
#pragma unroll
        for (int nt = 0; nt < 4; ++nt)
#pragma unroll
            for (int e = 0; e < 8; ++e) acc[mt][nt][e] = 0.0f;

    for (int k0 = 0; k0 < K; k0 += 32) {
        __syncthreads();
#pragma unroll
        for (int i = 0; i < 2; ++i) {
            int c   = tid + i * 256;           // 0..511, 16B chunks
            int row = c >> 2;
            int cc  = (c & 3) * 8;
            *(uint4*)&As[row * 32 + cc] =
                *(const uint4*)(A + (size_t)(blockM + row) * K + k0 + cc);
            *(uint4*)&Bs[row * 32 + cc] =
                *(const uint4*)(Bm + (size_t)(blockN + row) * K + k0 + cc);
            // prefetch next K-tile into cache (global_prefetch_b8)
            if (k0 + 32 < K) {
                __builtin_prefetch(A + (size_t)(blockM + row) * K + k0 + 32 + cc, 0, 1);
                __builtin_prefetch(Bm + (size_t)(blockN + row) * K + k0 + 32 + cc, 0, 1);
            }
        }
        __syncthreads();

        const int koA = hi16 * 16;    // A K half select: lane half h takes k 16h..16h+15
        const int koB = hi16 * 16;    // B K half select: same contiguous k order as A
        v16bf af[2];
#pragma unroll
        for (int mt = 0; mt < 2; ++mt)
            af[mt] = load_afrag(&As[(wm * 32 + mt * 16 + l15) * 32 + koA]);
#pragma unroll
        for (int nt = 0; nt < 4; ++nt) {
            v16bf bfrag = *(const v16bf*)&Bs[(wn * 64 + nt * 16 + l15) * 32 + koB];
#pragma unroll
            for (int mt = 0; mt < 2; ++mt)
                acc[mt][nt] = wmma_bf16(af[mt], bfrag, acc[mt][nt]);
        }
    }

    // C/D layout: VGPR e <-> row e (lanes 0-15) / e+8 (lanes 16-31), col = lane&15
    const int rsel = hi16 * 8;
#pragma unroll
    for (int mt = 0; mt < 2; ++mt)
#pragma unroll
        for (int nt = 0; nt < 4; ++nt)
#pragma unroll
            for (int e = 0; e < 8; ++e) {
                int gr = blockM + wm * 32 + mt * 16 + e + rsel;
                int gc = blockN + wn * 64 + nt * 16 + l15;
                float v = acc[mt][nt][e];
                if (F32OUT) {
                    ((float*)C)[(size_t)gr * Nn + gc] = v + bias[gc];
                } else {
                    ((bf16_t*)C)[(size_t)gr * Nn + gc] = f2bf(v);
                }
            }
}

// ---------------------------------------------------------------------------
// Per-(b,tok,h,half) L2 normalization of 64-elem rows; folds cos_scale into q.
// One wave per row; repacks into [b][h][half][tok][64].
// ---------------------------------------------------------------------------
__global__ __launch_bounds__(256)
void qk_norm_kernel(const bf16_t* __restrict__ Q, bf16_t* __restrict__ out,
                    const float* __restrict__ cos_head_delta,
                    const float* __restrict__ raw_p, int apply_scale) {
    const int lane = threadIdx.x & 31;
    const int rid  = blockIdx.x * (blockDim.x >> 5) + (threadIdx.x >> 5); // 0..65535
    const int tokb = rid >> 4;          // b*SEQ + tok
    const int r    = rid & 15;
    const int h    = r >> 1;
    const int half = r & 1;

    const bf16_t* src = Q + (size_t)rid * 64;   // contiguous head-half row
    float a = bf2f(src[2 * lane]);
    float b = bf2f(src[2 * lane + 1]);
    float ss = a * a + b * b;
#pragma unroll
    for (int m = 1; m < 32; m <<= 1) ss += __shfl_xor(ss, m, 32);
    float nrm = sqrtf(ss);

    float scale = 1.0f;
    if (apply_scale) {
        float raw    = raw_p[0];
        float gscale = 15.0f / (1.0f + __expf(-raw));   // COS_MIN=0, COS_MAX=15
        float mean   = 0.0f;
#pragma unroll
        for (int i = 0; i < HEADS; ++i) mean += cos_head_delta[i];
        mean *= (1.0f / HEADS);
        scale = gscale * (1.0f + 0.5f * fast_tanh(cos_head_delta[h] - mean));
    }
    float inv = scale / fmaxf(nrm, 1e-6f);

    const int bb  = tokb >> 11;
    const int tok = tokb & 2047;
    bf16_t* dst = out + (((size_t)(bb * HEADS + h) * 2 + half) * SEQ + tok) * 64;
    dst[2 * lane]     = f2bf(a * inv);
    dst[2 * lane + 1] = f2bf(b * inv);
}

// ---------------------------------------------------------------------------
// V repack: [b*SEQ+tok][h*128+vd] -> vT [b][h][vd][tok]
// ---------------------------------------------------------------------------
__global__ __launch_bounds__(256)
void v_repack_kernel(const bf16_t* __restrict__ V, bf16_t* __restrict__ vT) {
    int id  = blockIdx.x * 256 + threadIdx.x;   // 0..4194303
    int tok = id & 2047;
    int vd  = (id >> 11) & 127;
    int h   = (id >> 18) & 7;
    int b   = id >> 21;
    vT[id] = V[((size_t)(b * SEQ + tok)) * DIMC + h * DV + vd];
}

// ---------------------------------------------------------------------------
// Fused dual-softmax differential attention + RMS subln.
// Cosine attention => logits bounded by LOGIT_CAP, so we use a FIXED softmax
// offset: no running max, no accumulator rescaling, no per-iteration
// cross-lane reductions. Row sums accumulate per-lane, reduced once at end.
// Block: 128 thr (4 waves), each wave owns 16 query rows; grid (SEQ/64, B*H).
// ---------------------------------------------------------------------------
__global__ __launch_bounds__(128)
void attn_kernel(const bf16_t* __restrict__ qn, const bf16_t* __restrict__ kn,
                 const bf16_t* __restrict__ vT,
                 const float* __restrict__ lq1, const float* __restrict__ lk1,
                 const float* __restrict__ lq2, const float* __restrict__ lk2,
                 const float* __restrict__ delta_gain,
                 const float* __restrict__ subln,
                 bf16_t* __restrict__ Y) {
    __shared__ alignas(64) bf16_t k1s[32 * 64];        // [key][dh]
    __shared__ alignas(64) bf16_t k2s[32 * 64];
    __shared__ alignas(64) bf16_t vs[128 * 32];        // [vd][key]
    __shared__ alignas(64) bf16_t ps[4][2][16 * 32];   // per-wave P1/P2

    const int bh   = blockIdx.y;
    const int h    = bh & 7;
    const int tid  = threadIdx.x;
    const int wave = tid >> 5;
    const int lane = tid & 31;
    const int l15  = lane & 15;
    const int hi16 = lane >> 4;
    const int row0 = blockIdx.x * 64 + wave * 16;

    // lambda_full = exp(sum lq1*lk1) - exp(sum lq2*lk2) + 0.5
    float s1 = 0.0f, s2 = 0.0f;
#pragma unroll 8
    for (int i = 0; i < DHH; ++i) { s1 += lq1[i] * lk1[i]; s2 += lq2[i] * lk2[i]; }
    const float lambda_full = __expf(s1) - __expf(s2) + LAMBDA_INIT;
    const float gain = delta_gain[h];

    // Q fragments (cos_scale pre-folded by qk_norm)
    const bf16_t* q1b = qn + ((size_t)bh * 2 + 0) * SEQ * 64;
    const bf16_t* q2b = qn + ((size_t)bh * 2 + 1) * SEQ * 64;
    const int qrow = row0 + l15;
    const int koA  = hi16 * 16;
    const int koB  = hi16 * 16;
    v16bf qf1[2], qf2[2];
#pragma unroll
    for (int ks = 0; ks < 2; ++ks) {
        qf1[ks] = load_afrag(q1b + (size_t)qrow * 64 + ks * 32 + koA);
        qf2[ks] = load_afrag(q2b + (size_t)qrow * 64 + ks * 32 + koA);
    }

    v8f O1[8], O2[8];
    float l1a[8], l2a[8];   // per-lane partial row sums (this lane's columns)
#pragma unroll
    for (int nt = 0; nt < 8; ++nt)
#pragma unroll
        for (int e = 0; e < 8; ++e) { O1[nt][e] = 0.0f; O2[nt][e] = 0.0f; }
#pragma unroll
    for (int e = 0; e < 8; ++e) { l1a[e] = 0.0f; l2a[e] = 0.0f; }

    const bf16_t* k1b = kn + ((size_t)bh * 2 + 0) * SEQ * 64;
    const bf16_t* k2b = kn + ((size_t)bh * 2 + 1) * SEQ * 64;
    const bf16_t* vb  = vT + (size_t)bh * DV * SEQ;

    for (int j = 0; j < SEQ; j += 32) {
        __syncthreads();
        // Stage K1/K2 (32x64 each) and V^T (128x32) into LDS
#pragma unroll
        for (int i = 0; i < 2; ++i) {
            int c = tid + i * 128;              // 0..255 16B chunks
            int key = c >> 3, cc = (c & 7) * 8;
            *(uint4*)&k1s[key * 64 + cc] =
                *(const uint4*)(k1b + (size_t)(j + key) * 64 + cc);
            *(uint4*)&k2s[key * 64 + cc] =
                *(const uint4*)(k2b + (size_t)(j + key) * 64 + cc);
        }
#pragma unroll
        for (int i = 0; i < 4; ++i) {
            int c = tid + i * 128;              // 0..511
            int vd = c >> 2, cc = (c & 3) * 8;
            *(uint4*)&vs[vd * 32 + cc] =
                *(const uint4*)(vb + (size_t)vd * SEQ + j + cc);
        }
        __syncthreads();

        // L1/L2 scores: 16 rows x 32 keys
        v8f L1[2], L2[2];
#pragma unroll
        for (int nt = 0; nt < 2; ++nt)
#pragma unroll
            for (int e = 0; e < 8; ++e) { L1[nt][e] = 0.0f; L2[nt][e] = 0.0f; }
#pragma unroll
        for (int nt = 0; nt < 2; ++nt)
#pragma unroll
            for (int ks = 0; ks < 2; ++ks) {
                v16bf kf1 = *(const v16bf*)&k1s[(nt * 16 + l15) * 64 + ks * 32 + koB];
                v16bf kf2 = *(const v16bf*)&k2s[(nt * 16 + l15) * 64 + ks * 32 + koB];
                L1[nt] = wmma_bf16(qf1[ks], kf1, L1[nt]);
                L2[nt] = wmma_bf16(qf2[ks], kf2, L2[nt]);
            }

        // Gate + fixed-offset exponentials; accumulate per-lane row sums
#pragma unroll
        for (int nt = 0; nt < 2; ++nt)
#pragma unroll
            for (int e = 0; e < 8; ++e) {
                float a = L1[nt][e];
                float b = L2[nt][e];
                float g = fast_tanh(gain * (a - b));
                float p1 = __expf(a + g - LOGIT_CAP);
                float p2 = __expf(b - g - LOGIT_CAP);
                l1a[e] += p1;
                l2a[e] += p2;
                int pr = e + hi16 * 8;
                ps[wave][0][pr * 32 + nt * 16 + l15] = f2bf(p1);
                ps[wave][1][pr * 32 + nt * 16 + l15] = f2bf(p2);
            }
        __syncthreads();   // make P visible for transposed A-fragment reads

        // O += P * V
        v16bf pa1 = load_afrag(&ps[wave][0][l15 * 32 + koA]);
        v16bf pa2 = load_afrag(&ps[wave][1][l15 * 32 + koA]);
#pragma unroll
        for (int nt = 0; nt < 8; ++nt) {
            v16bf vf = *(const v16bf*)&vs[(nt * 16 + l15) * 32 + koB];
            O1[nt] = wmma_bf16(pa1, vf, O1[nt]);
            O2[nt] = wmma_bf16(pa2, vf, O2[nt]);
        }
    }

    // Reduce row sums across the 16-lane group (once), normalize, combine.
#pragma unroll
    for (int e = 0; e < 8; ++e) {
        float t1 = l1a[e], t2 = l2a[e];
#pragma unroll
        for (int m = 1; m < 16; m <<= 1) {
            t1 += __shfl_xor(t1, m, 32);
            t2 += __shfl_xor(t2, m, 32);
        }
        float i1 = 1.0f / t1;
        float i2 = lambda_full / t2;
#pragma unroll
        for (int nt = 0; nt < 8; ++nt)
            O1[nt][e] = O1[nt][e] * i1 - O2[nt][e] * i2;
    }
    // RMS over the 128-wide value dim, subln weight, *(1-LAMBDA_INIT)
    float rr[8];
#pragma unroll
    for (int e = 0; e < 8; ++e) {
        float ss = 0.0f;
#pragma unroll
        for (int nt = 0; nt < 8; ++nt) ss += O1[nt][e] * O1[nt][e];
#pragma unroll
        for (int m = 1; m < 16; m <<= 1) ss += __shfl_xor(ss, m, 32);
        rr[e] = rsqrtf(ss * (1.0f / 128.0f) + 1e-5f);
    }
    const int b = bh >> 3;
#pragma unroll
    for (int nt = 0; nt < 8; ++nt) {
        int col = nt * 16 + l15;
        float sw = subln[col] * (1.0f - LAMBDA_INIT);
#pragma unroll
        for (int e = 0; e < 8; ++e) {
            int gr = row0 + e + hi16 * 8;
            float yv = O1[nt][e] * rr[e] * sw;
            Y[((size_t)(b * SEQ + gr)) * DIMC + h * DV + col] = f2bf(yv);
        }
    }
}

// ---------------------------------------------------------------------------
// Host side
// ---------------------------------------------------------------------------
extern "C" void kernel_launch(void* const* d_in, const int* in_sizes, int n_in,
                              void* d_out, int out_size, void* d_ws, size_t ws_size,
                              hipStream_t stream) {
    (void)in_sizes; (void)n_in; (void)out_size; (void)ws_size;

    const float* x    = (const float*)d_in[0];
    const float* Wq   = (const float*)d_in[1];
    const float* Wk   = (const float*)d_in[2];
    const float* Wv   = (const float*)d_in[3];
    const float* Wo   = (const float*)d_in[4];
    const float* bo   = (const float*)d_in[5];
    const float* lq1  = (const float*)d_in[6];
    const float* lk1  = (const float*)d_in[7];
    const float* lq2  = (const float*)d_in[8];
    const float* lk2  = (const float*)d_in[9];
    const float* dg   = (const float*)d_in[10];
    const float* chd  = (const float*)d_in[11];
    const float* raw  = (const float*)d_in[12];
    const float* subw = (const float*)d_in[13];
    float* out = (float*)d_out;

    char* ws = (char*)d_ws;
    size_t off = 0;
    bf16_t* xb  = (bf16_t*)(ws + off); off += (size_t)MTOT * DIMC * 2;   // 8 MB
    bf16_t* Wqb = (bf16_t*)(ws + off); off += (size_t)DIMC * DIMC * 2;
    bf16_t* Wkb = (bf16_t*)(ws + off); off += (size_t)DIMC * DIMC * 2;
    bf16_t* Wvb = (bf16_t*)(ws + off); off += (size_t)DIMC * DIMC * 2;
    bf16_t* Wob = (bf16_t*)(ws + off); off += (size_t)DIMC * DIMC * 2;
    bf16_t* Qb  = (bf16_t*)(ws + off); off += (size_t)MTOT * DIMC * 2;
    bf16_t* Kb  = (bf16_t*)(ws + off); off += (size_t)MTOT * DIMC * 2;
    bf16_t* Vb  = (bf16_t*)(ws + off); off += (size_t)MTOT * DIMC * 2;
    bf16_t* qnb = (bf16_t*)(ws + off); off += (size_t)MTOT * DIMC * 2;
    bf16_t* knb = (bf16_t*)(ws + off); off += (size_t)MTOT * DIMC * 2;
    bf16_t* vTb = (bf16_t*)(ws + off); off += (size_t)MTOT * DIMC * 2;
    bf16_t* Yb  = (bf16_t*)(ws + off); off += (size_t)MTOT * DIMC * 2;

    const int NX = MTOT * DIMC;      // 4194304
    const int NW = DIMC * DIMC;      // 1048576
    f32_to_bf16_kernel<<<NX / 256, 256, 0, stream>>>(x,  xb,  NX);
    f32_to_bf16_kernel<<<NW / 256, 256, 0, stream>>>(Wq, Wqb, NW);
    f32_to_bf16_kernel<<<NW / 256, 256, 0, stream>>>(Wk, Wkb, NW);
    f32_to_bf16_kernel<<<NW / 256, 256, 0, stream>>>(Wv, Wvb, NW);
    f32_to_bf16_kernel<<<NW / 256, 256, 0, stream>>>(Wo, Wob, NW);

    dim3 gg(DIMC / 128, MTOT / 128);   // (8, 32)
    gemm_kernel<false><<<gg, 256, 0, stream>>>(xb, Wqb, (void*)Qb, nullptr, MTOT, DIMC, DIMC);
    gemm_kernel<false><<<gg, 256, 0, stream>>>(xb, Wkb, (void*)Kb, nullptr, MTOT, DIMC, DIMC);
    gemm_kernel<false><<<gg, 256, 0, stream>>>(xb, Wvb, (void*)Vb, nullptr, MTOT, DIMC, DIMC);

    // 65536 rows, 8 waves/block
    qk_norm_kernel<<<8192, 256, 0, stream>>>(Qb, qnb, chd, raw, 1);
    qk_norm_kernel<<<8192, 256, 0, stream>>>(Kb, knb, chd, raw, 0);
    v_repack_kernel<<<NX / 256, 256, 0, stream>>>(Vb, vTb);

    attn_kernel<<<dim3(SEQ / 64, BATCH * HEADS), 128, 0, stream>>>(
        qnb, knb, vTb, lq1, lk1, lq2, lk2, dg, subw, Yb);

    gemm_kernel<true><<<gg, 256, 0, stream>>>(Yb, Wob, (void*)out, bo, MTOT, DIMC, DIMC);
}
